// NeuralSparseAutoencoder_15874199126651
// MI455X (gfx1250) — matrix operand order repaired and checked by reading, not compiled, not run on hardware
//
#include <hip/hip_runtime.h>
#include <hip/hip_bf16.h>
#include <stdint.h>

// Problem constants (from reference)
#define B_ROWS  8192
#define D_IN    4096
#define D_HID   16384
#define K_TOP   64
#define ACT_CAP 128   // capacity for active list (>= K_TOP to absorb ties)

typedef __attribute__((ext_vector_type(16))) __bf16        bf16x16;
typedef __attribute__((ext_vector_type(8)))  float         f32x8;
typedef __attribute__((ext_vector_type(4)))  unsigned int  u32x4;
typedef __attribute__((ext_vector_type(4)))  int           i32x4;

union FragBF { bf16x16 v; u32x4 u[2]; };

// CDNA5 async global->LDS path (ASYNCcnt-tracked DMA), guarded so a missing
// builtin falls back to the synchronous load/ds_store path.
// Builtin signature (from probe diagnostics): (v4i AS1*, v4i AS3*, imm, imm).
#if __has_builtin(__builtin_amdgcn_global_load_async_to_lds_b128) && \
    __has_builtin(__builtin_amdgcn_s_wait_asynccnt)
#define USE_ASYNC_LDS 1
typedef __attribute__((address_space(1))) i32x4 as1_i32x4;
typedef __attribute__((address_space(3))) i32x4 as3_i32x4;
#else
#define USE_ASYNC_LDS 0
#endif

__device__ __forceinline__ unsigned short f32_to_bf16_rne(float f) {
  unsigned int u = __float_as_uint(f);
  unsigned int r = u + 0x7FFFu + ((u >> 16) & 1u);  // round-to-nearest-even
  return (unsigned short)(r >> 16);
}
// Monotonic float->uint transform (larger float => larger uint)
__device__ __forceinline__ unsigned int f2ord(float f) {
  unsigned int u = __float_as_uint(f);
  return (u & 0x80000000u) ? ~u : (u | 0x80000000u);
}
__device__ __forceinline__ float ord2f(unsigned int u) {
  unsigned int v = (u & 0x80000000u) ? (u & 0x7FFFFFFFu) : ~u;
  return __uint_as_float(v);
}

// ---------------------------------------------------------------------------
// Kernel 1: W f32 -> bf16 + per-row inverse L2 norm
// ---------------------------------------------------------------------------
__global__ __launch_bounds__(256)
void prep_w_kernel(const float* __restrict__ W,
                   unsigned short* __restrict__ wbf,
                   float* __restrict__ inv_norm) {
  __shared__ float red[256];
  const int row = blockIdx.x, tid = threadIdx.x;
  const float* src = W + (size_t)row * D_IN;
  unsigned short* dst = wbf + (size_t)row * D_IN;
  float ss = 0.f;
  for (int c = tid; c < D_IN; c += 256) {
    float v = src[c];
    ss += v * v;
    dst[c] = f32_to_bf16_rne(v);
  }
  red[tid] = ss;
  __syncthreads();
  for (int s = 128; s > 0; s >>= 1) {
    if (tid < s) red[tid] += red[tid + s];
    __syncthreads();
  }
  if (tid == 0) inv_norm[row] = 1.0f / sqrtf(red[0]);
}

// ---------------------------------------------------------------------------
// Kernel 2: x f32 -> bf16
// ---------------------------------------------------------------------------
__global__ __launch_bounds__(256)
void prep_x_kernel(const float* __restrict__ x, unsigned short* __restrict__ xbf) {
  const size_t base = (size_t)blockIdx.x * 4096;
  for (int c = threadIdx.x; c < 4096; c += 256)
    xbf[base + c] = f32_to_bf16_rne(x[base + c]);
}

// ---------------------------------------------------------------------------
// Kernel 3: encoder GEMM  encoded_pre = x @ W^T + b   (bf16 WMMA, f32 accum)
// Block tile 128(M) x 128(N), K-step 32, DOUBLE-BUFFERED LDS.
// 8 waves, each owns a 64x32 sub-tile = 4x2 grid of 16x16
// v_wmma_f32_16x16x32_bf16 accumulators. With async global->LDS the next
// K-slice's DMA overlaps the current slice's WMMAs; one barrier per K-step.
// ---------------------------------------------------------------------------
__device__ __forceinline__ void issue_tile_loads(
    const unsigned short* __restrict__ xbf,
    const unsigned short* __restrict__ wbf,
    unsigned short* As, unsigned short* Bs,
    int tid, int m0, int n0, int kk) {
#pragma unroll
  for (int i = 0; i < 2; i++) {
    const int l = tid + i * 256;           // 0..511
    const int r = l >> 2;                  // 0..127
    const int c = (l & 3) * 8;             // 0,8,16,24 halves (16B chunks)
    const unsigned short* ga = &xbf[(size_t)(m0 + r) * D_IN + kk + c];
    const unsigned short* gb = &wbf[(size_t)(n0 + r) * D_IN + kk + c];
#if USE_ASYNC_LDS
    __builtin_amdgcn_global_load_async_to_lds_b128(
        (as1_i32x4*)(void*)ga, (as3_i32x4*)(void*)&As[r * 32 + c], 0, 0);
    __builtin_amdgcn_global_load_async_to_lds_b128(
        (as1_i32x4*)(void*)gb, (as3_i32x4*)(void*)&Bs[r * 32 + c], 0, 0);
#else
    *(u32x4*)&As[r * 32 + c] = *(const u32x4*)ga;
    *(u32x4*)&Bs[r * 32 + c] = *(const u32x4*)gb;
#endif
  }
}

__global__ __launch_bounds__(256)
void encoder_gemm_kernel(const unsigned short* __restrict__ xbf,
                         const unsigned short* __restrict__ wbf,
                         const float* __restrict__ bias,
                         float* __restrict__ out_enc) {
  __shared__ unsigned short As[2][128 * 32];  // double-buffered, 2 x 8 KB
  __shared__ unsigned short Bs[2][128 * 32];

  const int tid  = threadIdx.x;
  const int lane = tid & 31;
  const int wave = tid >> 5;
  const int wm = wave >> 2;            // 0..1  (M direction)
  const int wn = wave & 3;             // 0..3  (N direction)
  const int m0 = blockIdx.y * 128;
  const int n0 = blockIdx.x * 128;
  const int half = lane >> 4;          // K-half selector per ISA layout
  const int l16  = lane & 15;

  f32x8 acc[4][2];
  for (int i = 0; i < 4; i++)
    for (int j = 0; j < 2; j++)
      for (int r = 0; r < 8; r++) acc[i][j][r] = 0.f;

  // Prologue: fill buffer 0
  issue_tile_loads(xbf, wbf, As[0], Bs[0], tid, m0, n0, 0);

  int buf = 0;
  for (int kk = 0; kk < D_IN; kk += 32) {
#if USE_ASYNC_LDS
    __builtin_amdgcn_s_wait_asynccnt(0);   // own async loads for As/Bs[buf] done
#endif
    __syncthreads();                       // all waves' loads landed; prev reads done
    if (kk + 32 < D_IN)                    // overlap next slice's DMA with WMMAs
      issue_tile_loads(xbf, wbf, As[buf ^ 1], Bs[buf ^ 1], tid, m0, n0, kk + 32);

    // Fragment loads per ISA 16-bit A layout:
    //   lane<16: K {0..7, 16..23}; lane>=16: K {8..15, 24..31}
    FragBF fa[4], fb[2];
#pragma unroll
    for (int mt = 0; mt < 4; mt++) {
      const int mrow = wm * 64 + mt * 16 + l16;
      const int c0 = half * 8;
      fa[mt].u[0] = *(const u32x4*)&As[buf][mrow * 32 + c0];
      fa[mt].u[1] = *(const u32x4*)&As[buf][mrow * 32 + c0 + 16];
    }
    // B (32x16, K-major source): lane<16 -> col n, K {0..7, 16..23}; lane>=16 -> K {8..15, 24..31}
#pragma unroll
    for (int nt = 0; nt < 2; nt++) {
      const int ncol = wn * 32 + nt * 16 + l16;
      const int k0 = half * 8;
      fb[nt].u[0] = *(const u32x4*)&Bs[buf][ncol * 32 + k0];
      fb[nt].u[1] = *(const u32x4*)&Bs[buf][ncol * 32 + k0 + 16];
    }
#pragma unroll
    for (int mt = 0; mt < 4; mt++)
#pragma unroll
      for (int nt = 0; nt < 2; nt++)
        acc[mt][nt] = __builtin_amdgcn_wmma_f32_16x16x32_bf16(
            false, fa[mt].v, false, fb[nt].v, (short)0, acc[mt][nt],
            false, false);
    buf ^= 1;
  }

  // Epilogue: bias add + store. C layout: n = lane%16, m = r + 8*(lane>>4)
#pragma unroll
  for (int nt = 0; nt < 2; nt++) {
    const int gn = n0 + wn * 32 + nt * 16 + l16;
    const float bv = bias[gn];
#pragma unroll
    for (int mt = 0; mt < 4; mt++) {
      const int gmb = m0 + wm * 64 + mt * 16 + half * 8;
#pragma unroll
      for (int r = 0; r < 8; r++)
        out_enc[(size_t)(gmb + r) * D_HID + gn] = acc[mt][nt][r] + bv;
    }
  }
}

// ---------------------------------------------------------------------------
// Kernel 4: exact top-k threshold via 4-round radix-select in LDS.
// Row (64 KB) staged once in LDS as ordered uints; CDNA5 has 320 KB/WGP.
// Then in-place where(>= thr) rewrite + index emission + active list.
// ---------------------------------------------------------------------------
__global__ __launch_bounds__(256)
void topk_select_kernel(float* __restrict__ enc,
                        int* __restrict__ out_idx,
                        const float* __restrict__ inv_norm,
                        int* __restrict__ act_idx,
                        float* __restrict__ act_val,
                        int* __restrict__ act_cnt) {
  extern __shared__ unsigned int sdyn[];
  unsigned int* vals = sdyn;            // D_HID ordered-uint keys
  unsigned int* hist = sdyn + D_HID;    // 256 bins
  __shared__ unsigned int s_prefix;
  __shared__ int s_kth;
  __shared__ int s_cnt;

  const int row = blockIdx.x, tid = threadIdx.x;
  float* rowp = enc + (size_t)row * D_HID;

  for (int i = tid; i < D_HID; i += 256) vals[i] = f2ord(rowp[i]);
  if (tid == 0) { s_prefix = 0u; s_kth = K_TOP; s_cnt = 0; }
  __syncthreads();

  const unsigned int masks[4] = {0x00000000u, 0xFF000000u, 0xFFFF0000u, 0xFFFFFF00u};
#pragma unroll
  for (int rnd = 0; rnd < 4; rnd++) {
    const int shift = 24 - rnd * 8;
    const unsigned int msk = masks[rnd];
    hist[tid] = 0u;                     // blockDim == 256 == #bins
    __syncthreads();
    const unsigned int pfx = s_prefix;
    for (int i = tid; i < D_HID; i += 256) {
      const unsigned int u = vals[i];
      if ((u & msk) == pfx) atomicAdd(&hist[(u >> shift) & 0xFFu], 1u);
    }
    __syncthreads();
    if (tid == 0) {
      int kth = s_kth;
      unsigned int cum = 0;
      for (int b = 255; b >= 0; --b) {
        const unsigned int c = hist[b];
        if (cum + c >= (unsigned int)kth) {
          s_prefix = pfx | (((unsigned int)b) << shift);
          s_kth = kth - (int)cum;
          break;
        }
        cum += c;
      }
    }
    __syncthreads();
  }

  const unsigned int thr = s_prefix;    // ordered-uint of exact k-th largest
  for (int i = tid; i < D_HID; i += 256) {
    const unsigned int u = vals[i];
    const bool keep = (u >= thr);       // matches reference's >= threshold
    const float orig = ord2f(u);
    rowp[i] = keep ? orig : 0.0f;
    if (keep) {
      const int slot = atomicAdd(&s_cnt, 1);
      if (slot < K_TOP) out_idx[(size_t)row * K_TOP + slot] = i;
      if (slot < ACT_CAP) {
        act_idx[row * ACT_CAP + slot] = i;
        act_val[row * ACT_CAP + slot] = orig * inv_norm[i];  // pre-scale by 1/||W_j||
      }
    }
  }
  __syncthreads();
  if (tid == 0) act_cnt[row] = (s_cnt < ACT_CAP) ? s_cnt : ACT_CAP;
}

// ---------------------------------------------------------------------------
// Kernel 5: sparse decode  decoded[i,:] = sum_j act_val * W_bf16[j,:]
// W_bf16 (128 MB) is L2-resident -> gathers run at L2 bandwidth.
// ---------------------------------------------------------------------------
__global__ __launch_bounds__(256)
void decode_kernel(const unsigned short* __restrict__ wbf,
                   const int* __restrict__ act_idx,
                   const float* __restrict__ act_val,
                   const int* __restrict__ act_cnt,
                   float* __restrict__ decoded) {
  __shared__ int aidx[ACT_CAP];
  __shared__ float aval[ACT_CAP];
  __shared__ int s_n;
  const int row = blockIdx.x, tid = threadIdx.x;
  if (tid == 0) s_n = act_cnt[row];
  __syncthreads();
  const int n = s_n;
  if (tid < n) {
    aidx[tid] = act_idx[row * ACT_CAP + tid];
    aval[tid] = act_val[row * ACT_CAP + tid];
  }
  __syncthreads();

  const int col0 = tid * 16;            // 256 threads * 16 cols = 4096
  float acc[16];
#pragma unroll
  for (int c = 0; c < 16; c++) acc[c] = 0.f;

  for (int e = 0; e < n; e++) {
    const int j = aidx[e];
    const float v = aval[e];
    const u32x4* wp = (const u32x4*)&wbf[(size_t)j * D_IN + col0];
    const u32x4 w0 = wp[0], w1 = wp[1];
#pragma unroll
    for (int q = 0; q < 4; q++) {
      const unsigned int p0 = w0[q], p1 = w1[q];
      acc[2 * q]         += v * __uint_as_float((p0 & 0xFFFFu) << 16);
      acc[2 * q + 1]     += v * __uint_as_float(p0 & 0xFFFF0000u);
      acc[8 + 2 * q]     += v * __uint_as_float((p1 & 0xFFFFu) << 16);
      acc[8 + 2 * q + 1] += v * __uint_as_float(p1 & 0xFFFF0000u);
    }
  }
  float* out = decoded + (size_t)row * D_IN + col0;
#pragma unroll
  for (int c = 0; c < 16; c++) out[c] = acc[c];
}

// ---------------------------------------------------------------------------
// Host-side launcher
// ---------------------------------------------------------------------------
extern "C" void kernel_launch(void* const* d_in, const int* in_sizes, int n_in,
                              void* d_out, int out_size, void* d_ws, size_t ws_size,
                              hipStream_t stream) {
  (void)in_sizes; (void)n_in; (void)out_size; (void)ws_size;
  const float* x = (const float*)d_in[0];   // (8192, 4096)
  const float* W = (const float*)d_in[1];   // (16384, 4096)
  const float* b = (const float*)d_in[2];   // (16384,)
  // d_in[3] is k == 64, fixed at compile time here.

  // Output layout: decoded | encoded | indices (int bits in float slots)
  float* out_decoded = (float*)d_out;
  float* out_encoded = out_decoded + (size_t)B_ROWS * D_IN;
  int*   out_indices = (int*)(out_encoded + (size_t)B_ROWS * D_HID);

  // Workspace layout
  char* ws = (char*)d_ws;
  unsigned short* wbf = (unsigned short*)ws;  ws += (size_t)D_HID * D_IN * 2;   // 128 MB
  float* inv_norm = (float*)ws;               ws += (size_t)D_HID * 4;
  unsigned short* xbf = (unsigned short*)ws;  ws += (size_t)B_ROWS * D_IN * 2;  // 64 MB
  int* act_cnt = (int*)ws;                    ws += (size_t)B_ROWS * 4;
  int* act_idx = (int*)ws;                    ws += (size_t)B_ROWS * ACT_CAP * 4;
  float* act_val = (float*)ws;

  prep_w_kernel<<<D_HID, 256, 0, stream>>>(W, wbf, inv_norm);
  prep_x_kernel<<<(B_ROWS * D_IN) / 4096, 256, 0, stream>>>(x, xbf);

  dim3 ggrid(D_HID / 128, B_ROWS / 128);  // (128, 64) blocks
  encoder_gemm_kernel<<<ggrid, 256, 0, stream>>>(xbf, wbf, b, out_encoded);

  const size_t smem = (size_t)(D_HID + 256) * sizeof(unsigned int);  // 66.5 KB
  topk_select_kernel<<<B_ROWS, 256, smem, stream>>>(out_encoded, out_indices,
                                                    inv_norm, act_idx, act_val,
                                                    act_cnt);
  decode_kernel<<<B_ROWS, 256, 0, stream>>>(wbf, act_idx, act_val, act_cnt,
                                            out_decoded);
}
